// MultiHeadSelfAttention_10084583211544
// MI455X (gfx1250) — matrix operand order repaired and checked by reading, not compiled, not run on hardware
//
#include <hip/hip_runtime.h>
#include <hip/hip_bf16.h>

// ---------------------------------------------------------------------------
// Multi-head self attention for MI455X (gfx1250), wave32 + WMMA f16->f32.
// B=2, S=2048, D=1024, H=16, dk=64.
// ---------------------------------------------------------------------------

typedef _Float16 h16 __attribute__((ext_vector_type(16)));
typedef _Float16 h8  __attribute__((ext_vector_type(8)));
typedef _Float16 h4  __attribute__((ext_vector_type(4)));
typedef float    f8  __attribute__((ext_vector_type(8)));
typedef float    f4  __attribute__((ext_vector_type(4)));

#define Bb  2
#define Ss  2048
#define Dd  1024
#define Hh  16
#define DK  64
#define ROWS (Bb * Ss)          // 4096

// ---- WMMA fragment loaders (layouts per CDNA5 ISA 7.12.2) ------------------

// A-matrix 16x32 f16: lane L holds row m=L&15; lanes 0-15 cover K=0-7 (v0-3)
// and K=16-23 (v4-7); lanes 16-31 cover K=8-15 and K=24-31.
static __device__ inline h16 load_a_frag(const _Float16* __restrict__ p0, int lda) {
    const int lane = threadIdx.x & 31;
    const int m  = lane & 15;
    const int kb = (lane >> 4) << 3;          // 0 or 8
    const _Float16* p = p0 + (size_t)m * lda + kb;
    h8 lo = *(const h8*)p;                    // K = kb .. kb+7
    h8 hi = *(const h8*)(p + 16);             // K = kb+16 .. kb+23
    return __builtin_shufflevector(lo, hi, 0,1,2,3,4,5,6,7,8,9,10,11,12,13,14,15);
}

// B-matrix 32x16 f16, source indexed [n][k] (row-major, stride ldb):
// lane L holds column n=L&15; lanes 0-15 hold K=0-7,16-23, lanes 16-31 K=8-15,24-31.
static __device__ inline h16 load_b_frag(const _Float16* __restrict__ p0, int ldb) {
    const int lane = threadIdx.x & 31;
    const int n  = lane & 15;
    const int kb = (lane >> 4) << 3;          // 0 or 8
    const _Float16* p = p0 + (size_t)n * ldb + kb;
    h8 lo = *(const h8*)p;
    h8 hi = *(const h8*)(p + 16);
    return __builtin_shufflevector(lo, hi, 0,1,2,3,4,5,6,7,8,9,10,11,12,13,14,15);
}

static __device__ inline f8 wmma32(h16 a, h16 b, f8 c) {
    return __builtin_amdgcn_wmma_f32_16x16x32_f16(false, a, false, b, (short)0, c,
                                                  false, false);
}

// ---- Kernel 1: fp32 -> f16 conversion (vectorized) -------------------------

__global__ __launch_bounds__(256) void cvt_kernel(
    const float* __restrict__ x,  const float* __restrict__ wq,
    const float* __restrict__ wk, const float* __restrict__ wv,
    const float* __restrict__ wo,
    _Float16* __restrict__ xh,  _Float16* __restrict__ wqh,
    _Float16* __restrict__ wkh, _Float16* __restrict__ wvh,
    _Float16* __restrict__ woh)
{
    const int NX4 = (ROWS * Dd) / 4;          // 1M float4's
    const int NW4 = (Dd * Dd) / 4;            // 256K float4's
    const int idx = blockIdx.x * blockDim.x + threadIdx.x;
    const int stride = gridDim.x * blockDim.x;
    for (int i = idx; i < NX4; i += stride) {
        f4 v = ((const f4*)x)[i];
        h4 o = { (_Float16)v.x, (_Float16)v.y, (_Float16)v.z, (_Float16)v.w };
        ((h4*)xh)[i] = o;
    }
    for (int i = idx; i < NW4; i += stride) {
        f4 a = ((const f4*)wq)[i];
        f4 b = ((const f4*)wk)[i];
        f4 c = ((const f4*)wv)[i];
        f4 d = ((const f4*)wo)[i];
        ((h4*)wqh)[i] = h4{ (_Float16)a.x, (_Float16)a.y, (_Float16)a.z, (_Float16)a.w };
        ((h4*)wkh)[i] = h4{ (_Float16)b.x, (_Float16)b.y, (_Float16)b.z, (_Float16)b.w };
        ((h4*)wvh)[i] = h4{ (_Float16)c.x, (_Float16)c.y, (_Float16)c.z, (_Float16)c.w };
        ((h4*)woh)[i] = h4{ (_Float16)d.x, (_Float16)d.y, (_Float16)d.z, (_Float16)d.w };
    }
}

// ---- Kernel 2: fused QKV projection ----------------------------------------
// One wave computes a 32(row) x 64(col) tile of one of Q/K/V, double-buffered.
// Q scaled by 1/sqrt(dk)=0.125 at store. V stored transposed [B,H,dk,S].

__global__ __launch_bounds__(256) void qkv_kernel(
    const _Float16* __restrict__ xh,
    const _Float16* __restrict__ Wqh, const _Float16* __restrict__ Wkh,
    const _Float16* __restrict__ Wvh,
    _Float16* __restrict__ Qh, _Float16* __restrict__ Kh,
    _Float16* __restrict__ Vth)
{
    const int wid = blockIdx.x * 8 + (threadIdx.x >> 5);
    const int cg  = wid & 15;                 // 64-col group == head index
    const int rt  = (wid >> 4) & 127;         // 32-row tile (ROWS/32)
    const int mat = wid >> 11;                // 0=Q 1=K 2=V
    const int row0 = rt << 5;

    const _Float16* W = (mat == 0) ? Wqh : (mat == 1) ? Wkh : Wvh;
    const _Float16* Wbase = W + (size_t)(cg * DK) * Dd;
    const _Float16* Abase = xh + (size_t)row0 * Dd;

    f8 c00 = {}, c01 = {}, c02 = {}, c03 = {};   // rows 0..15
    f8 c10 = {}, c11 = {}, c12 = {}, c13 = {};   // rows 16..31

    h16 a0 = load_a_frag(Abase, Dd);
    h16 a1 = load_a_frag(Abase + 16 * Dd, Dd);
    h16 b0 = load_b_frag(Wbase +  0 * Dd, Dd);
    h16 b1 = load_b_frag(Wbase + 16 * Dd, Dd);
    h16 b2 = load_b_frag(Wbase + 32 * Dd, Dd);
    h16 b3 = load_b_frag(Wbase + 48 * Dd, Dd);

    for (int kk = 32; kk <= Dd; kk += 32) {
        h16 na0, na1, nb0, nb1, nb2, nb3;
        if (kk < Dd) {                         // prefetch next K-step
            na0 = load_a_frag(Abase + kk, Dd);
            na1 = load_a_frag(Abase + 16 * Dd + kk, Dd);
            nb0 = load_b_frag(Wbase +  0 * Dd + kk, Dd);
            nb1 = load_b_frag(Wbase + 16 * Dd + kk, Dd);
            nb2 = load_b_frag(Wbase + 32 * Dd + kk, Dd);
            nb3 = load_b_frag(Wbase + 48 * Dd + kk, Dd);
        }
        c00 = wmma32(a0, b0, c00);  c10 = wmma32(a1, b0, c10);
        c01 = wmma32(a0, b1, c01);  c11 = wmma32(a1, b1, c11);
        c02 = wmma32(a0, b2, c02);  c12 = wmma32(a1, b2, c12);
        c03 = wmma32(a0, b3, c03);  c13 = wmma32(a1, b3, c13);
        a0 = na0; a1 = na1; b0 = nb0; b1 = nb1; b2 = nb2; b3 = nb3;
    }

    const int lane = threadIdx.x & 31;
    const int n  = lane & 15;
    const int mb = (lane >> 4) << 3;
    const int b  = row0 >> 11;                // / S
    const int s  = row0 & (Ss - 1);
    const int h  = cg;
    f8 cs[2][4] = { {c00, c01, c02, c03}, {c10, c11, c12, c13} };

    if (mat < 2) {
        const float scale = (mat == 0) ? 0.125f : 1.0f;
        _Float16* dst = ((mat == 0) ? Qh : Kh)
                        + ((size_t)(b * Hh + h) * Ss + s) * DK;
        #pragma unroll
        for (int r = 0; r < 2; ++r)
            #pragma unroll
            for (int j = 0; j < 4; ++j)
                #pragma unroll
                for (int i = 0; i < 8; ++i)
                    dst[(size_t)(r * 16 + mb + i) * DK + j * 16 + n] =
                        (_Float16)(cs[r][j][i] * scale);
    } else {
        // V^T: [dk, S]; s-dim contiguous -> 16-byte vector stores
        _Float16* dst = Vth + (size_t)(b * Hh + h) * DK * Ss + s;
        #pragma unroll
        for (int r = 0; r < 2; ++r)
            #pragma unroll
            for (int j = 0; j < 4; ++j) {
                h8 v;
                #pragma unroll
                for (int i = 0; i < 8; ++i) v[i] = (_Float16)cs[r][j][i];
                *(h8*)(dst + (size_t)(j * 16 + n) * Ss + r * 16 + mb) = v;
            }
    }
}

// ---- Kernel 3: causal flash attention --------------------------------------
// One wave handles a 16-row query tile for one (b,h); key tiles of 32.
// V fragments prefetched behind softmax; next K tile prefetched behind P@V.

__global__ __launch_bounds__(256) void attn_kernel(
    const _Float16* __restrict__ Qh, const _Float16* __restrict__ Kh,
    const _Float16* __restrict__ Vth, _Float16* __restrict__ Ah)
{
    __shared__ _Float16 Pbuf[8][16][32];      // per-wave P tile (C->A relayout)

    const int wid = blockIdx.x * 8 + (threadIdx.x >> 5);
    const int qt = wid & 127;                 // S/16
    const int h  = (wid >> 7) & 15;
    const int b  = wid >> 11;
    const int q0 = qt << 4;

    const int lane = threadIdx.x & 31;
    const int nn = lane & 15;
    const int mb = (lane >> 4) << 3;
    const int w  = threadIdx.x >> 5;

    const size_t bh = (size_t)(b * Hh + h);
    const _Float16* Qbase = Qh  + (bh * Ss + q0) * DK;
    const _Float16* Kbase = Kh  + bh * Ss * DK;
    const _Float16* Vbase = Vth + bh * DK * Ss;

    const h16 aq0 = load_a_frag(Qbase, DK);       // Q rows, K-dim 0..31
    const h16 aq1 = load_a_frag(Qbase + 32, DK);  // K-dim 32..63

    f8 acc0 = {}, acc1 = {}, acc2 = {}, acc3 = {};
    float mrow[8], lrow[8];
    #pragma unroll
    for (int i = 0; i < 8; ++i) { mrow[i] = -__builtin_inff(); lrow[i] = 0.f; }

    const int ntiles = (q0 + 16 + 31) >> 5;

    // preload K fragments for tile 0
    h16 kb0 = load_b_frag(Kbase, DK);
    h16 kb1 = load_b_frag(Kbase + 32, DK);
    h16 kb2 = load_b_frag(Kbase + 16 * DK, DK);
    h16 kb3 = load_b_frag(Kbase + 16 * DK + 32, DK);

    for (int t = 0; t < ntiles; ++t) {
        const int k0 = t << 5;

        // scores: s0 = keys k0..k0+15, s1 = keys k0+16..k0+31 (Q pre-scaled)
        f8 s0 = {}, s1 = {};
        s0 = wmma32(aq0, kb0, s0);
        s0 = wmma32(aq1, kb1, s0);
        s1 = wmma32(aq0, kb2, s1);
        s1 = wmma32(aq1, kb3, s1);

        // issue V loads now; consumed only after softmax (latency hidden)
        const _Float16* Vb = Vbase + k0;
        h16 vb0 = load_b_frag(Vb +  0 * Ss, Ss);
        h16 vb1 = load_b_frag(Vb + 16 * Ss, Ss);
        h16 vb2 = load_b_frag(Vb + 32 * Ss, Ss);
        h16 vb3 = load_b_frag(Vb + 48 * Ss, Ss);

        // issue next tile's K loads (clamped; dead value on last iter)
        const int k0n = (t + 1 < ntiles) ? (k0 + 32) : k0;
        const _Float16* Kbn = Kbase + (size_t)k0n * DK;
        h16 nkb0 = load_b_frag(Kbn, DK);
        h16 nkb1 = load_b_frag(Kbn + 32, DK);
        h16 nkb2 = load_b_frag(Kbn + 16 * DK, DK);
        h16 nkb3 = load_b_frag(Kbn + 16 * DK + 32, DK);

        // causal mask (only tiles touching the diagonal)
        if (k0 + 31 > q0) {
            #pragma unroll
            for (int i = 0; i < 8; ++i) {
                const int q = q0 + mb + i;
                if (k0 + nn > q)      s0[i] = -__builtin_inff();
                if (k0 + 16 + nn > q) s1[i] = -__builtin_inff();
            }
        }

        // online softmax: row max over 16 lanes of each half-wave
        float rmax[8];
        #pragma unroll
        for (int i = 0; i < 8; ++i) rmax[i] = fmaxf(s0[i], s1[i]);
        #pragma unroll
        for (int i = 0; i < 8; ++i)
            #pragma unroll
            for (int d = 1; d < 16; d <<= 1)
                rmax[i] = fmaxf(rmax[i], __shfl_xor(rmax[i], d, 32));

        float alpha[8], p0[8], p1[8], rsum[8];
        #pragma unroll
        for (int i = 0; i < 8; ++i) {
            const float mnew = fmaxf(mrow[i], rmax[i]);
            alpha[i] = __expf(mrow[i] - mnew);
            mrow[i]  = mnew;
            p0[i] = __expf(s0[i] - mnew);
            p1[i] = __expf(s1[i] - mnew);
            rsum[i] = p0[i] + p1[i];
        }
        #pragma unroll
        for (int i = 0; i < 8; ++i)
            #pragma unroll
            for (int d = 1; d < 16; d <<= 1)
                rsum[i] += __shfl_xor(rsum[i], d, 32);
        #pragma unroll
        for (int i = 0; i < 8; ++i) {
            lrow[i] = lrow[i] * alpha[i] + rsum[i];
            acc0[i] *= alpha[i]; acc1[i] *= alpha[i];
            acc2[i] *= alpha[i]; acc3[i] *= alpha[i];
        }

        // P tile (C layout) -> LDS -> A layout
        #pragma unroll
        for (int i = 0; i < 8; ++i) {
            Pbuf[w][mb + i][nn]      = (_Float16)p0[i];
            Pbuf[w][mb + i][16 + nn] = (_Float16)p1[i];
        }
        __builtin_amdgcn_wave_barrier();      // keep ds_store -> ds_load order
        const h16 ap = load_a_frag(&Pbuf[w][0][0], 32);

        // acc += P @ V
        acc0 = wmma32(ap, vb0, acc0);
        acc1 = wmma32(ap, vb1, acc1);
        acc2 = wmma32(ap, vb2, acc2);
        acc3 = wmma32(ap, vb3, acc3);

        kb0 = nkb0; kb1 = nkb1; kb2 = nkb2; kb3 = nkb3;
    }

    // epilogue: normalize and store f16 attention output [B,S,D]
    _Float16* dst = Ah + ((size_t)(b * Ss + q0)) * Dd + h * DK;
    f8 accs[4] = {acc0, acc1, acc2, acc3};
    #pragma unroll
    for (int j = 0; j < 4; ++j)
        #pragma unroll
        for (int i = 0; i < 8; ++i)
            dst[(size_t)(mb + i) * Dd + j * 16 + nn] =
                (_Float16)(accs[j][i] / lrow[i]);
}

// ---- Kernel 4: output projection (fp32 result), 32x64 tiles ----------------

__global__ __launch_bounds__(256) void oproj_kernel(
    const _Float16* __restrict__ Ahh, const _Float16* __restrict__ Woh,
    float* __restrict__ out)
{
    const int wid = blockIdx.x * 8 + (threadIdx.x >> 5);
    const int cg = wid & 15;
    const int rt = wid >> 4;                  // 0..127
    const int row0 = rt << 5;

    const _Float16* Abase = Ahh + (size_t)row0 * Dd;
    const _Float16* Wbase = Woh + (size_t)(cg * DK) * Dd;

    f8 c00 = {}, c01 = {}, c02 = {}, c03 = {};
    f8 c10 = {}, c11 = {}, c12 = {}, c13 = {};

    h16 a0 = load_a_frag(Abase, Dd);
    h16 a1 = load_a_frag(Abase + 16 * Dd, Dd);
    h16 b0 = load_b_frag(Wbase +  0 * Dd, Dd);
    h16 b1 = load_b_frag(Wbase + 16 * Dd, Dd);
    h16 b2 = load_b_frag(Wbase + 32 * Dd, Dd);
    h16 b3 = load_b_frag(Wbase + 48 * Dd, Dd);

    for (int kk = 32; kk <= Dd; kk += 32) {
        h16 na0, na1, nb0, nb1, nb2, nb3;
        if (kk < Dd) {
            na0 = load_a_frag(Abase + kk, Dd);
            na1 = load_a_frag(Abase + 16 * Dd + kk, Dd);
            nb0 = load_b_frag(Wbase +  0 * Dd + kk, Dd);
            nb1 = load_b_frag(Wbase + 16 * Dd + kk, Dd);
            nb2 = load_b_frag(Wbase + 32 * Dd + kk, Dd);
            nb3 = load_b_frag(Wbase + 48 * Dd + kk, Dd);
        }
        c00 = wmma32(a0, b0, c00);  c10 = wmma32(a1, b0, c10);
        c01 = wmma32(a0, b1, c01);  c11 = wmma32(a1, b1, c11);
        c02 = wmma32(a0, b2, c02);  c12 = wmma32(a1, b2, c12);
        c03 = wmma32(a0, b3, c03);  c13 = wmma32(a1, b3, c13);
        a0 = na0; a1 = na1; b0 = nb0; b1 = nb1; b2 = nb2; b3 = nb3;
    }

    const int lane = threadIdx.x & 31;
    const int n  = lane & 15;
    const int mb = (lane >> 4) << 3;
    f8 cs[2][4] = { {c00, c01, c02, c03}, {c10, c11, c12, c13} };
    float* dst = out + (size_t)row0 * Dd + cg * DK;
    #pragma unroll
    for (int r = 0; r < 2; ++r)
        #pragma unroll
        for (int j = 0; j < 4; ++j)
            #pragma unroll
            for (int i = 0; i < 8; ++i)
                dst[(size_t)(r * 16 + mb + i) * Dd + j * 16 + n] = cs[r][j][i];
}

// ---- launch ----------------------------------------------------------------

extern "C" void kernel_launch(void* const* d_in, const int* in_sizes, int n_in,
                              void* d_out, int out_size, void* d_ws, size_t ws_size,
                              hipStream_t stream)
{
    (void)in_sizes; (void)n_in; (void)out_size; (void)ws_size;

    const float* x  = (const float*)d_in[0];
    const float* Wq = (const float*)d_in[1];
    const float* Wk = (const float*)d_in[2];
    const float* Wv = (const float*)d_in[3];
    const float* Wo = (const float*)d_in[4];
    float* out = (float*)d_out;

    char* ws = (char*)d_ws;
    const size_t MB = 1u << 20;
    _Float16* xh  = (_Float16*)(ws);               //  8 MB: x f16
    _Float16* wqh = (_Float16*)(ws +  8 * MB);     //  2 MB
    _Float16* wkh = (_Float16*)(ws + 10 * MB);     //  2 MB
    _Float16* wvh = (_Float16*)(ws + 12 * MB);     //  2 MB
    _Float16* woh = (_Float16*)(ws + 14 * MB);     //  2 MB
    _Float16* Qh  = (_Float16*)(ws + 16 * MB);     //  8 MB [B,H,S,dk], pre-scaled
    _Float16* Kh  = (_Float16*)(ws + 24 * MB);     //  8 MB [B,H,S,dk]
    _Float16* Vth = (_Float16*)(ws + 32 * MB);     //  8 MB [B,H,dk,S]
    _Float16* Ah  = (_Float16*)(ws + 40 * MB);     //  8 MB [B,S,D] f16

    cvt_kernel<<<2048, 256, 0, stream>>>(x, Wq, Wk, Wv, Wo,
                                         xh, wqh, wkh, wvh, woh);
    // 3 mats * 128 row-tiles * 16 col-groups = 6144 waves / 8 per block
    qkv_kernel<<<768, 256, 0, stream>>>(xh, wqh, wkh, wvh, Qh, Kh, Vth);
    // B*H*(S/16) = 4096 waves
    attn_kernel<<<512, 256, 0, stream>>>(Qh, Kh, Vth, Ah);
    // 128 row-tiles * 16 col-groups = 2048 waves
    oproj_kernel<<<256, 256, 0, stream>>>(Ah, woh, out);
}
